// UpdateBlock_34265249088339
// MI455X (gfx1250) — matrix operand order repaired and checked by reading, not compiled, not run on hardware
//
#include <hip/hip_runtime.h>
#include <hip/hip_bf16.h>
#include <math.h>

#define BB 2
#define NN 32768
#define KK 16

typedef __bf16 bf16_t;
typedef __attribute__((ext_vector_type(16))) __bf16 v16bf;
typedef __attribute__((ext_vector_type(8)))  float  v8f;
typedef __attribute__((ext_vector_type(16))) unsigned short v16u;

// -------- precomputed weight-fragment area offsets (in ushorts) --------
// each fragment = 32 lanes * 16 bf16 = 512 ush
#define OFF_CORR 0        //  8 frags (4 mt x 2 ks)
#define OFF_FLOW 4096     //  4 frags (4 x 1)
#define OFF_ME   6144     // 16 frags (4 x 4)
#define OFF_Z    14336    // 24 frags (4 x 6)
#define OFF_R    26624    // 24
#define OFF_Q    38912    // 24
#define OFF_FH1  51200    //  8
#define OFF_SC1  55296    // 12 frags (4 x 3)
#define OFF_SC2  61440    //  8
#define OFF_SC3  65536    //  8
#define OFF_OUT1 69632    // 16 frags (4 x 4)
#define OFF_OUT2 77824    //  2 frags (1 x 2)
#define N_FRAGS_TOTAL 154

// native f32 -> bf16 (single v_cvt on gfx1250), returned as raw bits
__device__ __forceinline__ unsigned short f2bf_bits(float f) {
  union { bf16_t b; unsigned short s; } v; v.b = (bf16_t)f; return v.s;
}
__device__ __forceinline__ float bf2f(unsigned short s) {
  union { unsigned u; float f; } v; v.u = ((unsigned)s) << 16;
  return v.f;
}
__device__ __forceinline__ v16bf as_v16bf(v16u u) {
  union { v16u u; v16bf b; } c; c.u = u; return c.b;
}
__device__ __forceinline__ void wbar() { __builtin_amdgcn_wave_barrier(); }
__device__ __forceinline__ float sigmoidf_(float x) { return 1.0f / (1.0f + __expf(-x)); }
__device__ __forceinline__ float tanhf_(float x) {
  const float e = __expf(2.0f * x);
  return (e - 1.0f) / (e + 1.0f);
}

// pack 8 f32 -> 8 bf16 (one b128 store's worth)
__device__ __forceinline__ uint4 pack8(const float* f) {
  union { unsigned short s[8]; uint4 q; } u;
#pragma unroll
  for (int j = 0; j < 8; ++j) u.s[j] = f2bf_bits(f[j]);
  return u.q;
}
__device__ __forceinline__ void load8(const unsigned short* src, float* f) {
  union { unsigned short s[8]; uint4 q; } u;
  u.q = *(const uint4*)src;
#pragma unroll
  for (int j = 0; j < 8; ++j) f[j] = bf2f(u.s[j]);
}

// B (32x16 bf16, K x N): lane n = lane&15, K range = (lane>>4)*16 + j
__device__ __forceinline__ v16bf frag_b_lds(const unsigned short* act, int pitch,
                                            int kbase, int lane) {
  const int n  = lane & 15;
  const int hf = lane >> 4;
  return as_v16bf(*(const v16u*)(act + n * pitch + kbase + hf * 16));
}
// A fragment from pre-laid-out workspace: one 32B global load per lane
__device__ __forceinline__ v16bf frag_a_pre(const unsigned short* wf, int idx, int lane) {
  return as_v16bf(*(const v16u*)(wf + ((size_t)(idx * 32 + lane)) * 16));
}

// 64-out-channel GEMM tile over a wave's 16 points: 4 M-tiles x KSTEPS K-steps.
template <int KSTEPS>
__device__ __forceinline__ void gemm64p(const unsigned short* wf,
                                        const unsigned short* act, int pitch,
                                        int lane, v8f acc[4]) {
#pragma unroll
  for (int mt = 0; mt < 4; ++mt) {
    v8f c = {0.f, 0.f, 0.f, 0.f, 0.f, 0.f, 0.f, 0.f};
#pragma unroll
    for (int ks = 0; ks < KSTEPS; ++ks) {
      v16bf a = frag_a_pre(wf, mt * KSTEPS + ks, lane);
      v16bf b = frag_b_lds(act, pitch, ks * 32, lane);
      c = __builtin_amdgcn_wmma_f32_16x16x32_bf16(false, a, false, b, (short)0, c,
                                                  false, false);
    }
    acc[mt] = c;
  }
}

// ============ Kernel 0: lay weights out as per-lane WMMA A fragments ============
// A (16x32 bf16, M x K): lane m = lane&15, half = lane>>4,
// element j -> K = 16*half + j   (contiguous, matches frag_b_lds)
__global__ __launch_bounds__(32)
void k0_prep(const float* __restrict__ w_corr, const float* __restrict__ w_flow,
             const float* __restrict__ w_me,   const float* __restrict__ wz,
             const float* __restrict__ wr,     const float* __restrict__ wq,
             const float* __restrict__ w_fh1,  const float* __restrict__ w_sc1,
             const float* __restrict__ w_sc2,  const float* __restrict__ w_sc3,
             const float* __restrict__ w_out1, const float* __restrict__ w_out2,
             unsigned short* __restrict__ wfrag) {
  struct WD { const float* W; int cols; int rows; int ksteps; int nfrags; int off; };
  const WD wd[12] = {
    {w_corr, 64, 64, 2,  8, OFF_CORR}, {w_flow,   3, 64, 1,  4, OFF_FLOW},
    {w_me,  128, 61, 4, 16, OFF_ME},   {wz,     192, 64, 6, 24, OFF_Z},
    {wr,    192, 64, 6, 24, OFF_R},    {wq,     192, 64, 6, 24, OFF_Q},
    {w_fh1,  64, 64, 2,  8, OFF_FH1},  {w_sc1,   67, 64, 3, 12, OFF_SC1},
    {w_sc2,  64, 64, 2,  8, OFF_SC2},  {w_sc3,   64, 64, 2,  8, OFF_SC3},
    {w_out1,128, 64, 4, 16, OFF_OUT1}, {w_out2,  64,  3, 2,  2, OFF_OUT2},
  };
  int fid = blockIdx.x;
  int wi = 0;
  while (wi < 11 && fid >= wd[wi].nfrags) { fid -= wd[wi].nfrags; ++wi; }
  const WD& w = wd[wi];
  const int mt = fid / w.ksteps, ks = fid % w.ksteps;
  const int lane = threadIdx.x & 31;
  const int m  = mt * 16 + (lane & 15);
  const int hf = lane >> 4;
  unsigned short* dst = wfrag + w.off + ((size_t)((mt * w.ksteps + ks) * 32 + lane)) * 16;
#pragma unroll
  for (int j = 0; j < 16; ++j) {
    const int k = ks * 32 + 16 * hf + j;
    const float v = (m < w.rows && k < w.cols) ? w.W[(size_t)m * w.cols + k] : 0.0f;
    dst[j] = f2bf_bits(v);
  }
}

// stage a (64ch x 16pt) fp32 channel-major global tile into LDS [pt][ch] bf16
__device__ __forceinline__ void stage64(const float* __restrict__ src, size_t row0,
                                        int n0, unsigned short* dst, int pitch,
                                        int coff, int lane) {
#pragma unroll
  for (int it = 0; it < 8; ++it) {
    const int t = lane + it * 32;          // 0..255 -> 64ch x 4 float4
    const int c = t >> 2, i0 = (t & 3) * 4;
    const float4 v = *(const float4*)&src[(row0 + c) * NN + n0 + i0];
    dst[(i0 + 0) * pitch + coff + c] = f2bf_bits(v.x);
    dst[(i0 + 1) * pitch + coff + c] = f2bf_bits(v.y);
    dst[(i0 + 2) * pitch + coff + c] = f2bf_bits(v.z);
    dst[(i0 + 3) * pitch + coff + c] = f2bf_bits(v.w);
  }
}

// ================= Kernel 1: MotionEncoder + ConvGRU + o1 =================
__global__ __launch_bounds__(128)
void k1_update(const float* __restrict__ net, const float* __restrict__ inp,
               const float* __restrict__ corr, const float* __restrict__ flow,
               const float* __restrict__ b_corr, const float* __restrict__ b_flow,
               const float* __restrict__ b_me,
               const float* __restrict__ bz, const float* __restrict__ br,
               const float* __restrict__ bq, const float* __restrict__ b_fh1,
               const unsigned short* __restrict__ wfrag,
               float* __restrict__ net_out,
               unsigned short* __restrict__ netT,
               unsigned short* __restrict__ o1ws) {
  __shared__ __align__(32) unsigned short s_hx[4][16 * 192];
  __shared__ __align__(32) unsigned short s_mf[4][16 * 128];
  __shared__ __align__(32) unsigned short s_fp[4][16 * 32];

  const int lane = threadIdx.x & 31;
  const int wv   = threadIdx.x >> 5;
  const int g    = blockIdx.x * 4 + wv;
  const int b    = g / (NN / 16);
  const int n0   = (g % (NN / 16)) * 16;

  unsigned short* hx = s_hx[wv];
  unsigned short* mf = s_mf[wv];
  unsigned short* fp = s_fp[wv];

  const int nloc = lane & 15, hf = lane >> 4;
  v8f acc[4];
  float t8[8];

  // ---- cor = relu(w_corr @ corr + b) -> mf[:,0:64] (hx used as corr staging)
  stage64(corr, (size_t)b * 64, n0, hx, 192, 0, lane);
  wbar();
  gemm64p<2>(wfrag + OFF_CORR, hx, 192, lane, acc);
#pragma unroll
  for (int mt = 0; mt < 4; ++mt) {
    const int c0 = mt * 16 + 8 * hf;
#pragma unroll
    for (int r = 0; r < 8; ++r) t8[r] = fmaxf(acc[mt][r] + b_corr[c0 + r], 0.f);
    *(uint4*)&mf[nloc * 128 + c0] = pack8(t8);
  }

  // ---- flo = relu(w_flow @ flow + b) -> mf[:,64:128] (flow padded 3->32)
  for (int t = lane; t < 16 * 32; t += 32) fp[t] = 0;
  wbar();
  for (int t = lane; t < 16 * 3; t += 32) {
    const int i = t / 3, j = t % 3;
    fp[i * 32 + j] = f2bf_bits(flow[((size_t)b * NN + n0 + i) * 3 + j]);
  }
  wbar();
  gemm64p<1>(wfrag + OFF_FLOW, fp, 32, lane, acc);
#pragma unroll
  for (int mt = 0; mt < 4; ++mt) {
    const int c0 = mt * 16 + 8 * hf;
#pragma unroll
    for (int r = 0; r < 8; ++r) t8[r] = fmaxf(acc[mt][r] + b_flow[c0 + r], 0.f);
    *(uint4*)&mf[nloc * 128 + 64 + c0] = pack8(t8);
  }
  wbar();

  // ---- mf61 = relu(w_me @ [cor;flo] + b) -> hx[:,128:189]; flow -> hx[:,189:192]
  gemm64p<4>(wfrag + OFF_ME, mf, 128, lane, acc);
#pragma unroll
  for (int mt = 0; mt < 4; ++mt) {
    const int c0 = mt * 16 + 8 * hf;
    if (c0 + 8 <= 61) {
#pragma unroll
      for (int r = 0; r < 8; ++r) t8[r] = fmaxf(acc[mt][r] + b_me[c0 + r], 0.f);
      *(uint4*)&hx[nloc * 192 + 128 + c0] = pack8(t8);
    } else if (c0 < 61) {
#pragma unroll
      for (int r = 0; r < 8; ++r)
        if (c0 + r < 61)
          hx[nloc * 192 + 128 + c0 + r] =
              f2bf_bits(fmaxf(acc[mt][r] + b_me[c0 + r], 0.f));
    }
  }
  for (int t = lane; t < 48; t += 32) {
    const int i = t / 3, j = t % 3;
    hx[i * 192 + 189 + j] = fp[i * 32 + j];
  }
  // ---- hx[:,0:64] = net, hx[:,64:128] = inp
  stage64(net, (size_t)b * 64, n0, hx, 192, 0, lane);
  stage64(inp, (size_t)b * 64, n0, hx, 192, 64, lane);
  wbar();

  // ---- z / r gates ----
  float zv[4][8], netv[4][8];
  gemm64p<6>(wfrag + OFF_Z, hx, 192, lane, acc);
#pragma unroll
  for (int mt = 0; mt < 4; ++mt) {
    const int c0 = mt * 16 + 8 * hf;
#pragma unroll
    for (int r = 0; r < 8; ++r) zv[mt][r] = sigmoidf_(acc[mt][r] + bz[c0 + r]);
  }
  gemm64p<6>(wfrag + OFF_R, hx, 192, lane, acc);
  wbar();
#pragma unroll
  for (int mt = 0; mt < 4; ++mt) {
    const int c0 = mt * 16 + 8 * hf;
    float nf[8];
    load8(&hx[nloc * 192 + c0], nf);
#pragma unroll
    for (int r = 0; r < 8; ++r) {
      const float rr = sigmoidf_(acc[mt][r] + br[c0 + r]);
      netv[mt][r] = nf[r];
      t8[r] = rr * nf[r];
    }
    *(uint4*)&hx[nloc * 192 + c0] = pack8(t8);   // hx[:,0:64] := r*net
  }
  wbar();

  // ---- q gate + net_new ----
  gemm64p<6>(wfrag + OFF_Q, hx, 192, lane, acc);
  const int n = n0 + nloc;
#pragma unroll
  for (int mt = 0; mt < 4; ++mt) {
    const int c0 = mt * 16 + 8 * hf;
#pragma unroll
    for (int r = 0; r < 8; ++r) {
      const float q  = tanhf_(acc[mt][r] + bq[c0 + r]);
      const float nn = (1.f - zv[mt][r]) * netv[mt][r] + zv[mt][r] * q;
      net_out[((size_t)b * 64 + c0 + r) * NN + n] = nn;
      t8[r] = nn;
    }
    const uint4 q4 = pack8(t8);
    *(uint4*)&hx[nloc * 192 + c0] = q4;                         // := net_new
    *(uint4*)&netT[((size_t)b * NN + n) * 64 + c0] = q4;
  }
  wbar();

  // ---- o1 = w_fh1 @ net_new + b (no relu) -> workspace bf16
  gemm64p<2>(wfrag + OFF_FH1, hx, 192, lane, acc);
#pragma unroll
  for (int mt = 0; mt < 4; ++mt) {
    const int c0 = mt * 16 + 8 * hf;
#pragma unroll
    for (int r = 0; r < 8; ++r) t8[r] = acc[mt][r] + b_fh1[c0 + r];
    *(uint4*)&o1ws[((size_t)b * NN + n) * 64 + c0] = pack8(t8);
  }
}

// ================= Kernel 2: setconv + output head =================
__global__ __launch_bounds__(128)
void k2_flowhead(const int* __restrict__ edges, const float* __restrict__ edge_feats,
                 const float* __restrict__ b_sc1, const float* __restrict__ b_sc2,
                 const float* __restrict__ b_sc3, const float* __restrict__ b_out1,
                 const float* __restrict__ b_out2,
                 const unsigned short* __restrict__ wfrag,
                 const unsigned short* __restrict__ netT,
                 const unsigned short* __restrict__ o1ws,
                 float* __restrict__ delta_out) {
  __shared__ __align__(32) unsigned short s_nbr[4][16 * 96];
  __shared__ __align__(32) unsigned short s_sv[4][16 * 64];
  __shared__ __align__(32) unsigned short s_s2[4][16 * 64];
  __shared__ __align__(32) unsigned short s_oin[4][16 * 128];

  const int lane = threadIdx.x & 31;
  const int wv   = threadIdx.x >> 5;
  const int g    = blockIdx.x * 4 + wv;
  const int b    = g / (NN / 16);
  const int n0   = (g % (NN / 16)) * 16;

  unsigned short* nbr = s_nbr[wv];
  unsigned short* sv  = s_sv[wv];
  unsigned short* s2  = s_s2[wv];
  unsigned short* oin = s_oin[wv];
  const int nloc = lane & 15, hf = lane >> 4;
  v8f acc[4];
  float t8[8];

  // ---- stage A: per point, gather K=16 neighbors -> 67ch, conv, relu, maxpool
  const int k    = lane & 15;
  const int part = lane >> 4;   // each neighbor row filled by 2 lanes (64B each)
  for (int p = 0; p < 16; ++p) {
    const int n = n0 + p;
    const int e = edges[((size_t)b * NN + n) * KK + k];
    const unsigned short* src = netT + ((size_t)b * NN + e) * 64 + part * 32;
    *(v16u*)(nbr + k * 96 + part * 32)      = *(const v16u*)(src);
    *(v16u*)(nbr + k * 96 + part * 32 + 16) = *(const v16u*)(src + 16);
    if (part == 0) {
      const float* ef = edge_feats + (((size_t)b * NN + n) * KK + k) * 3;
      nbr[k * 96 + 64] = f2bf_bits(ef[0]);
      nbr[k * 96 + 65] = f2bf_bits(ef[1]);
      nbr[k * 96 + 66] = f2bf_bits(ef[2]);
#pragma unroll
      for (int c = 67; c < 96; ++c) nbr[k * 96 + c] = 0;  // K pad 67->96
    }
    wbar();
    gemm64p<3>(wfrag + OFF_SC1, nbr, 96, lane, acc);
#pragma unroll
    for (int mt = 0; mt < 4; ++mt) {
      const int c0 = mt * 16 + 8 * hf;
#pragma unroll
      for (int r = 0; r < 8; ++r) {
        float v = fmaxf(acc[mt][r] + b_sc1[c0 + r], 0.f);
        v = fmaxf(v, __shfl_xor(v, 1, 32));   // max over the 16 neighbor columns
        v = fmaxf(v, __shfl_xor(v, 2, 32));
        v = fmaxf(v, __shfl_xor(v, 4, 32));
        v = fmaxf(v, __shfl_xor(v, 8, 32));
        t8[r] = v;
      }
      if (nloc == 0) *(uint4*)&sv[p * 64 + c0] = pack8(t8);
    }
    wbar();
  }

  // ---- o1 -> oin[:,64:128]  (b128 copies)
#pragma unroll
  for (int it = 0; it < 4; ++it) {
    const int t = lane + it * 32;        // 128 chunks of 8 ush
    const int p = t >> 3, c0 = (t & 7) * 8;
    *(uint4*)&oin[p * 128 + 64 + c0] =
        *(const uint4*)&o1ws[((size_t)b * NN + n0 + p) * 64 + c0];
  }
  wbar();

  // ---- s2 = relu(w_sc2 @ sv + b)
  gemm64p<2>(wfrag + OFF_SC2, sv, 64, lane, acc);
#pragma unroll
  for (int mt = 0; mt < 4; ++mt) {
    const int c0 = mt * 16 + 8 * hf;
#pragma unroll
    for (int r = 0; r < 8; ++r) t8[r] = fmaxf(acc[mt][r] + b_sc2[c0 + r], 0.f);
    *(uint4*)&s2[nloc * 64 + c0] = pack8(t8);
  }
  wbar();

  // ---- s3 = relu(w_sc3 @ s2 + b) -> oin[:,0:64]
  gemm64p<2>(wfrag + OFF_SC3, s2, 64, lane, acc);
#pragma unroll
  for (int mt = 0; mt < 4; ++mt) {
    const int c0 = mt * 16 + 8 * hf;
#pragma unroll
    for (int r = 0; r < 8; ++r) t8[r] = fmaxf(acc[mt][r] + b_sc3[c0 + r], 0.f);
    *(uint4*)&oin[nloc * 128 + c0] = pack8(t8);
  }
  wbar();

  // ---- o = relu(w_out1 @ [s3;o1] + b) -> reuse s2 buffer
  gemm64p<4>(wfrag + OFF_OUT1, oin, 128, lane, acc);
#pragma unroll
  for (int mt = 0; mt < 4; ++mt) {
    const int c0 = mt * 16 + 8 * hf;
#pragma unroll
    for (int r = 0; r < 8; ++r) t8[r] = fmaxf(acc[mt][r] + b_out1[c0 + r], 0.f);
    *(uint4*)&s2[nloc * 64 + c0] = pack8(t8);
  }
  wbar();

  // ---- delta = w_out2(3x64, padded to 16 rows) @ o + b : 1 M-tile, 2 K-steps
  v8f d = {0.f, 0.f, 0.f, 0.f, 0.f, 0.f, 0.f, 0.f};
#pragma unroll
  for (int ks = 0; ks < 2; ++ks) {
    v16bf a  = frag_a_pre(wfrag + OFF_OUT2, ks, lane);
    v16bf bb = frag_b_lds(s2, 64, ks * 32, lane);
    d = __builtin_amdgcn_wmma_f32_16x16x32_bf16(false, a, false, bb, (short)0, d,
                                                false, false);
  }
#pragma unroll
  for (int r = 0; r < 8; ++r) {
    const int c = r + 8 * hf;
    if (c < 3)
      delta_out[((size_t)b * NN + n0 + nloc) * 3 + c] = d[r] + b_out2[c];
  }
}

// ================= host launcher =================
extern "C" void kernel_launch(void* const* d_in, const int* in_sizes, int n_in,
                              void* d_out, int out_size, void* d_ws, size_t ws_size,
                              hipStream_t stream) {
  (void)in_sizes; (void)n_in; (void)out_size; (void)ws_size;
  const float* net        = (const float*)d_in[0];
  const float* inp        = (const float*)d_in[1];
  const float* corr       = (const float*)d_in[2];
  const float* flow       = (const float*)d_in[3];
  const int*   edges      = (const int*)d_in[4];
  const float* edge_feats = (const float*)d_in[5];
  const float* w_corr = (const float*)d_in[6],  *b_corr = (const float*)d_in[7];
  const float* w_flow = (const float*)d_in[8],  *b_flow = (const float*)d_in[9];
  const float* w_me   = (const float*)d_in[10], *b_me   = (const float*)d_in[11];
  const float* wz     = (const float*)d_in[12], *bz     = (const float*)d_in[13];
  const float* wr     = (const float*)d_in[14], *br     = (const float*)d_in[15];
  const float* wq     = (const float*)d_in[16], *bq     = (const float*)d_in[17];
  const float* w_fh1  = (const float*)d_in[18], *b_fh1  = (const float*)d_in[19];
  const float* w_sc1  = (const float*)d_in[20], *b_sc1  = (const float*)d_in[21];
  const float* w_sc2  = (const float*)d_in[22], *b_sc2  = (const float*)d_in[23];
  const float* w_sc3  = (const float*)d_in[24], *b_sc3  = (const float*)d_in[25];
  const float* w_out1 = (const float*)d_in[26], *b_out1 = (const float*)d_in[27];
  const float* w_out2 = (const float*)d_in[28], *b_out2 = (const float*)d_in[29];

  float* net_out   = (float*)d_out;                         // (B,64,N)
  float* delta_out = (float*)d_out + (size_t)BB * 64 * NN;  // (B,N,3)

  unsigned short* netT  = (unsigned short*)d_ws;            // bf16 (B,N,64)
  unsigned short* o1ws  = netT + (size_t)BB * NN * 64;      // bf16 (B,N,64)
  unsigned short* wfrag = o1ws + (size_t)BB * NN * 64;      // bf16 A-fragments

  // 0) lay all weights out in WMMA A-fragment order (bf16, zero-padded)
  k0_prep<<<dim3(N_FRAGS_TOTAL), dim3(32), 0, stream>>>(
      w_corr, w_flow, w_me, wz, wr, wq, w_fh1, w_sc1, w_sc2, w_sc3, w_out1,
      w_out2, wfrag);

  const int n_wave_tiles = (BB * NN) / 16;   // 4096
  dim3 grid(n_wave_tiles / 4);               // 4 waves (128 thr) per block
  dim3 block(128);

  // 1) MotionEncoder + ConvGRU + o1
  k1_update<<<grid, block, 0, stream>>>(net, inp, corr, flow,
                                        b_corr, b_flow, b_me, bz, br, bq, b_fh1,
                                        wfrag, net_out, netT, o1ws);

  // 2) setconv (WMMA over 16 neighbors) + output head
  k2_flowhead<<<grid, block, 0, stream>>>(edges, edge_feats,
                                          b_sc1, b_sc2, b_sc3, b_out1, b_out2,
                                          wfrag, netT, o1ws, delta_out);
}
